// UpConvsample2d_22771916604155
// MI455X (gfx1250) — matrix operand order repaired and checked by reading, not compiled, not run on hardware
//
#include <hip/hip_runtime.h>
#include <hip/hip_bf16.h>
#include <stdint.h>

// ---------------------------------------------------------------------------
// StyleGAN upsample2x-conv3x3-FIR, split on-chip:
//   stage 0: NCHW f32 -> padded NHWC bf16 (enables async/TDM LDS staging)
//   stage 1 (WMMA): y = transposed-conv(x, w), sub-pixel decomposed
//                   -> 4 phases with 4/2/2/1 taps (avg 2.25) over 64x64 input
//   stage 2 (VALU): o = separable FIR3x3(y) from LDS, cropped to 128x128
// ~193 GFLOP bf16 WMMA (~77us @ 2.5 PFLOPS dense bf16); y stays in WGP LDS.
// Slab staging modes: 2 = TDM tensor_load_to_lds (1 DMA/slab, TENSORcnt),
// 1 = GLOBAL_LOAD_ASYNC_TO_LDS_B128 (ASYNCcnt), 0 = f32 fallback (small ws).
// ---------------------------------------------------------------------------

typedef __attribute__((ext_vector_type(16))) __bf16          bf16x16;
typedef __attribute__((ext_vector_type(16))) unsigned short  u16x16;
typedef __attribute__((ext_vector_type(4)))  unsigned short  u16x4;
typedef __attribute__((ext_vector_type(8)))  float           v8f;
typedef __attribute__((ext_vector_type(4)))  unsigned        u32x4;
typedef __attribute__((ext_vector_type(8)))  unsigned        u32x8;

union Frag32B { u16x16 u; bf16x16 b; };

__device__ __forceinline__ unsigned short f2bf(float f) {
  unsigned u = __float_as_uint(f);
  unsigned r = u + 0x7FFFu + ((u >> 16) & 1u);   // RNE
  return (unsigned short)(r >> 16);
}
__device__ __forceinline__ float bf2f(unsigned short b) {
  return __uint_as_float(((unsigned)b) << 16);
}

#define CIN    512
#define COUT   256
#define HW_IN  64
#define HW_OUT 128

// ---- ws layout --------------------------------------------------------------
#define WS_W_ELEMS (9u*16u*16u*32u*16u)            // 1,179,648 bf16 weights
#define XT_OFF     ((size_t)4u * 1024u * 1024u)    // 4 MB aligned
#define XT_ELEMS   ((size_t)16 * 66 * 66 * 512)    // padded NHWC bf16
#define XT_BYTES   (XT_ELEMS * 2u)                 // 71.4 MB

// ---------------------------------------------------------------------------
// Kernel A: phase-decomposed weights, pre-swizzled to A-fragment lane layout.
// ---------------------------------------------------------------------------
__global__ void phase_weights_kernel(const float* __restrict__ w,
                                     unsigned short* __restrict__ ww) {
  unsigned tid = blockIdx.x * blockDim.x + threadIdx.x;
  if (tid >= WS_W_ELEMS) return;
  unsigned e    = tid & 15u;
  unsigned lane = (tid >> 4) & 31u;
  unsigned co16 = (tid >> 9) & 15u;
  unsigned kb   = (tid >> 13) & 15u;
  unsigned ti   = tid >> 17;

  int co = (int)(co16 * 16u + (lane & 15u));
  int K  = (int)(e + ((lane >> 4) ? 16u : 0u));
  int ci = (int)(kb * 32u) + K;

  int pr, pc, dr, dc;
  if (ti < 4u)      { pr = 0; pc = 0; dr = (int)(ti >> 1); dc = (int)(ti & 1u); }
  else if (ti < 6u) { pr = 0; pc = 1; dr = (int)ti - 4;    dc = 0; }
  else if (ti < 8u) { pr = 1; pc = 0; dr = 0;              dc = (int)ti - 6; }
  else              { pr = 1; pc = 1; dr = 0;              dc = 0; }
  int u = pr ? 1 : 2 * dr;
  int v = pc ? 1 : 2 * dc;
  float W = w[((ci * COUT + co) * 3 + (2 - u)) * 3 + (2 - v)];
  ww[tid] = f2bf(W);
}

// ---------------------------------------------------------------------------
// Kernel A2: NCHW f32 -> padded NHWC bf16.  xt[n][prow 66][pcol 66][ci 512],
// prow=r+2, pcol=c+1, zero apron (staging needs no bounds checks).
// ---------------------------------------------------------------------------
__global__ void __launch_bounds__(256)
nhwc_convert_kernel(const float* __restrict__ x,
                    unsigned short* __restrict__ xt) {
  __shared__ unsigned short tile[64][66];
  unsigned bid  = blockIdx.x;
  unsigned prow = bid % 66u;
  unsigned n    = bid / 66u;
  unsigned tid  = threadIdx.x;
  unsigned short* rowp = xt + ((size_t)n * 66u + prow) * 66u * 512u;
  int r = (int)prow - 2;
  if (r < 0 || r >= HW_IN) {          // apron row
    uint4 z = make_uint4(0, 0, 0, 0);
    for (unsigned t = tid; t < 4224u; t += 256u)
      *reinterpret_cast<uint4*>(rowp + (size_t)t * 8u) = z;
    return;
  }
  for (unsigned t = tid; t < 128u; t += 256u) {   // apron cols 0 and 65
    unsigned pcol = (t >= 64u) ? 65u : 0u;
    *reinterpret_cast<uint4*>(rowp + (size_t)pcol * 512u + (t & 63u) * 8u) =
        make_uint4(0, 0, 0, 0);
  }
  const float* xr = x + (size_t)n * CIN * HW_IN * HW_IN +
                    (size_t)(unsigned)r * HW_IN;
  for (unsigned cb = 0; cb < 8u; ++cb) {
    __syncthreads();
    for (unsigned k = 0; k < 16u; ++k) {          // coalesced over columns
      unsigned idx = k * 256u + tid;
      unsigned ci_l = idx >> 6, c = idx & 63u;
      tile[ci_l][c] =
          f2bf(xr[(size_t)(cb * 64u + ci_l) * (HW_IN * HW_IN) + c]);
    }
    __syncthreads();
    for (unsigned k = 0; k < 16u; ++k) {          // coalesced over channels
      unsigned idx = k * 256u + tid;
      unsigned c = idx >> 6, ci_l = idx & 63u;
      rowp[(size_t)(c + 1u) * 512u + cb * 64u + ci_l] = tile[ci_l][c];
    }
  }
}

// ---------------------------------------------------------------------------
union SMem {
  unsigned short xs[2][6 * 66 * 32];     // 2 x 25,344 B
  unsigned short ys[10 * 16 * 132];      // 42,240 B
};                                        // union: 50,688 B -> 6 WGs/WGP

template <int TC>
__device__ __forceinline__ void compute_kb(const unsigned short* __restrict__ wkb,
                                           const unsigned short* xsb,
                                           unsigned bt, unsigned lane,
                                           v8f acc[10]) {
  unsigned nl = lane & 15u;
  unsigned hi = (lane >> 4) ? 16u : 0u;
#pragma unroll
  for (int ar = 0; ar < 5; ++ar) {
#pragma unroll
    for (int pr = 0; pr < 2; ++pr) {
      const int ri = 2 * ar + pr;
#pragma unroll
      for (int dr = 0; dr < (pr ? 1 : 2); ++dr) {
        unsigned xrow = pr ? (unsigned)(ar + 1) : (unsigned)(ar + dr);
#pragma unroll
        for (int dc = 0; dc < TC; ++dc) {
          int ti = (TC == 2) ? (pr ? (6 + dc) : (dr * 2 + dc))
                             : (pr ? 8 : (4 + dr));
          Frag32B a, b;
          a.u = *reinterpret_cast<const u16x16*>(wkb + (size_t)ti * 131072u);
          unsigned coloff = (TC == 2) ? (bt * 16u + nl + (unsigned)dc)
                                      : (bt * 16u + nl + 1u);
          b.u = *reinterpret_cast<const u16x16*>(
              xsb + (xrow * 66u + coloff) * 32u + hi);
          acc[ri] = __builtin_amdgcn_wmma_f32_16x16x32_bf16(
              false, a.b, false, b.b, (short)0, acc[ri], false, false);
        }
      }
    }
  }
}

__device__ __forceinline__ void wait_async_zero() {
#if __has_builtin(__builtin_amdgcn_s_wait_asynccnt)
  __builtin_amdgcn_s_wait_asynccnt(0);
#else
  asm volatile("s_wait_asynccnt 0" ::: "memory");
#endif
}

// TDM: one DMA for the whole 3D slab (32ci x 66col x 6row, 2B elems) from the
// padded NHWC tensor into LDS.  D# per ISA 8.3-8.6 (4 SGPR groups).
__device__ __forceinline__ void tdm_load_slab(const unsigned short* g,
                                              unsigned lds_byte) {
  unsigned long long ga = (unsigned long long)(size_t)g;
  u32x4 g0;
  g0[0] = 1u;                                            // count=1, user D#
  g0[1] = lds_byte;                                      // lds_addr
  g0[2] = (unsigned)(ga & 0xFFFFFFFFull);                // global_addr[31:0]
  g0[3] = (unsigned)((ga >> 32) & 0x01FFFFFFull) | (2u << 30); // [56:32]|type=2
  u32x8 g1;
  g1[0] = 1u << 16;                                      // data_size=1 (2B)
  g1[1] = (512u & 0xFFFFu) << 16;                        // tensor_dim0[15:0]
  g1[2] = (512u >> 16) | (66u << 16);                    // dim0[31:16]|dim1 lo
  g1[3] = 0u | (32u << 16);                              // dim1[31:16]|tile_dim0
  g1[4] = 66u | (6u << 16);                              // tile_dim1|tile_dim2
  g1[5] = 512u;                                          // dim0_stride[31:0]
  g1[6] = 0u | ((33792u & 0xFFFFu) << 16);               // d0s[47:32]|d1s[15:0]
  g1[7] = 33792u >> 16;                                  // dim1_stride[47:16]
  u32x4 g2;
  g2[0] = 66u;                                           // tensor_dim2
  g2[1] = 0u;
  g2[2] = 2230272u;                                      // dim2_stride (unused)
  g2[3] = 0u;                                            // |tile_dim3=0
  u32x4 g3 = (u32x4){0u, 0u, 0u, 0u};
  asm volatile("tensor_load_to_lds %0, %1, %2, %3"
               :: "s"(g0), "s"(g1), "s"(g2), "s"(g3) : "memory");
}

// MODE: 0 = f32 direct staging, 1 = async b128 copies, 2 = TDM
template <int MODE>
__global__ void __launch_bounds__(256)
subpix_conv_fir_kernel(const float* __restrict__ x,
                       const unsigned short* __restrict__ xt,
                       const unsigned short* __restrict__ ww,
                       float* __restrict__ out) {
  __shared__ SMem smem;

  unsigned bid    = blockIdx.x;
  unsigned co16   = bid & 15u;
  unsigned rowblk = (bid >> 4) & 15u;
  unsigned n      = bid >> 8;
  unsigned a0     = rowblk * 4u;

  unsigned tid  = threadIdx.x;
  unsigned lane = tid & 31u;
  unsigned wave = tid >> 5;
  unsigned pc = (unsigned)__builtin_amdgcn_readfirstlane((int)(wave >> 2));
  unsigned bt = (unsigned)__builtin_amdgcn_readfirstlane((int)(wave & 3u));

  const float* xn = x + (size_t)n * CIN * HW_IN * HW_IN;
  const unsigned short* xtn =
      (MODE != 0) ? (xt + ((size_t)n * 66u + a0) * 66u * 512u) : nullptr;
  unsigned ldsb[2];
  ldsb[0] = (unsigned)__builtin_amdgcn_readfirstlane(
      (int)(unsigned)(size_t)&smem.xs[0][0]);
  ldsb[1] = (unsigned)__builtin_amdgcn_readfirstlane(
      (int)(unsigned)(size_t)&smem.xs[1][0]);

  v8f acc[10];
#pragma unroll
  for (int i = 0; i < 10; ++i) acc[i] = (v8f){0, 0, 0, 0, 0, 0, 0, 0};

  const unsigned short* wbase = ww + (size_t)co16 * 512u + (size_t)lane * 16u;

  auto stage_f32 = [&](unsigned kb, unsigned buf) {
    for (unsigned t = tid; t < 3168u; t += 256u) {
      unsigned cq  = t / 396u;
      unsigned rc  = t - cq * 396u;
      unsigned row = rc / 66u;
      unsigned col = rc - row * 66u;
      int r_in = (int)a0 - 2 + (int)row;
      int c_in = (int)col - 1;
      float v0 = 0.f, v1 = 0.f, v2 = 0.f, v3 = 0.f;
      if (r_in >= 0 && r_in < HW_IN && c_in >= 0 && c_in < HW_IN) {
        const float* p = xn +
            ((size_t)(kb * 32u + cq * 4u) * HW_IN + (unsigned)r_in) * HW_IN +
            (unsigned)c_in;
        v0 = p[0]; v1 = p[4096]; v2 = p[8192]; v3 = p[12288];
      }
      uint2 pk;
      pk.x = (unsigned)f2bf(v0) | ((unsigned)f2bf(v1) << 16);
      pk.y = (unsigned)f2bf(v2) | ((unsigned)f2bf(v3) << 16);
      *reinterpret_cast<uint2*>(
          &smem.xs[buf][(row * 66u + col) * 32u + cq * 4u]) = pk;
    }
  };
  auto stage_async = [&](unsigned kb, unsigned buf) {
    const unsigned short* base = xtn + kb * 32u;
    for (unsigned t = tid; t < 1584u; t += 256u) {
      unsigned oct = t & 3u;
      unsigned rc  = t >> 2;
      unsigned col = rc % 66u;
      unsigned row = rc / 66u;
      const unsigned short* g = base + (size_t)(row * 66u + col) * 512u + oct * 8u;
      unsigned ldsa = (unsigned)(size_t)
          &smem.xs[buf][(row * 66u + col) * 32u + oct * 8u];
      unsigned long long ga = (unsigned long long)(size_t)g;
      asm volatile("global_load_async_to_lds_b128 %0, %1, off"
                   :: "v"(ldsa), "v"(ga) : "memory");
    }
  };
  auto stage_any = [&](unsigned kb, unsigned buf) {
    if (MODE == 0) stage_f32(kb, buf);
    else if (MODE == 1) stage_async(kb, buf);
    else if (wave == 0u) tdm_load_slab(xtn + kb * 32u, ldsb[buf]);
  };
  auto stage_wait = [&]() {
    if (MODE == 1) wait_async_zero();
    else if (MODE == 2) {
      if (wave == 0u) __builtin_amdgcn_s_wait_tensorcnt(0);
    }
  };

  stage_any(0u, 0u);
  stage_wait();
  __syncthreads();
  for (unsigned kb = 0; kb < 16u; ++kb) {
    if (kb + 1u < 16u) stage_any(kb + 1u, (kb + 1u) & 1u);
    const unsigned short* wkb = wbase + (size_t)kb * 8192u;
    __builtin_prefetch(wkb + 8192u, 0, 1);
    if (pc == 0u) compute_kb<2>(wkb, smem.xs[kb & 1u], bt, lane, acc);
    else          compute_kb<1>(wkb, smem.xs[kb & 1u], bt, lane, acc);
    stage_wait();
    __syncthreads();
  }

  {   // park y tile in LDS (x buffers dead -> union reuse OK)
    for (unsigned u = tid; u < 320u; u += 256u) {
      unsigned p = u >> 5, rem = u & 31u;
      smem.ys[(p * 16u + (rem >> 1)) * 132u + (rem & 1u)] = 0;
    }
    unsigned nl = lane & 15u, hiw = lane >> 4;
    unsigned qs = 2u * (bt * 16u + nl) + pc + 2u;
#pragma unroll
    for (int ar = 0; ar < 5; ++ar)
#pragma unroll
      for (int pr = 0; pr < 2; ++pr) {
        int ri = 2 * ar + pr;
        unsigned p = 2u * (unsigned)ar + (unsigned)pr;
#pragma unroll
        for (int v = 0; v < 8; ++v) {
          unsigned m = (unsigned)v + hiw * 8u;
          smem.ys[(p * 16u + m) * 132u + qs] = f2bf(acc[ri][v]);
        }
      }
  }
  __syncthreads();

  // separable FIR (k' = {0.4,1.2,0.4}), rolling 3-row b64 window loads
  unsigned co_l  = tid >> 4;
  unsigned jg    = tid & 15u;
  unsigned jbase = jg * 8u;
  float* outp = out + ((size_t)n * COUT + co16 * 16u + co_l) * HW_OUT * HW_OUT;

  float W[3][10];
  auto loadrow = [&](int p, float* dst) {
    const unsigned short* rp =
        &smem.ys[((unsigned)p * 16u + co_l) * 132u + jbase];
    u16x4 a = *reinterpret_cast<const u16x4*>(rp);
    u16x4 b = *reinterpret_cast<const u16x4*>(rp + 4);
    u16x4 c = *reinterpret_cast<const u16x4*>(rp + 8);
    dst[0] = bf2f(a[0]); dst[1] = bf2f(a[1]); dst[2] = bf2f(a[2]);
    dst[3] = bf2f(a[3]); dst[4] = bf2f(b[0]); dst[5] = bf2f(b[1]);
    dst[6] = bf2f(b[2]); dst[7] = bf2f(b[3]); dst[8] = bf2f(c[0]);
    dst[9] = bf2f(c[1]);
  };
  loadrow(0, W[0]);
  loadrow(1, W[1]);
#pragma unroll
  for (int ii = 0; ii < 8; ++ii) {
    loadrow(ii + 2, W[(ii + 2) % 3]);
    float V[10];
#pragma unroll
    for (int q = 0; q < 10; ++q)
      V[q] = 0.4f * (W[ii % 3][q] + W[(ii + 2) % 3][q]) +
             1.2f * W[(ii + 1) % 3][q];
    float o[8];
#pragma unroll
    for (int c = 0; c < 8; ++c)
      o[c] = 0.4f * (V[c] + V[c + 2]) + 1.2f * V[c + 1];
    unsigned i = rowblk * 8u + (unsigned)ii;
    float4 s0 = make_float4(o[0], o[1], o[2], o[3]);
    float4 s1 = make_float4(o[4], o[5], o[6], o[7]);
    *reinterpret_cast<float4*>(outp + (size_t)i * HW_OUT + jbase)      = s0;
    *reinterpret_cast<float4*>(outp + (size_t)i * HW_OUT + jbase + 4u) = s1;
  }
}

// ---------------------------------------------------------------------------
extern "C" void kernel_launch(void* const* d_in, const int* in_sizes, int n_in,
                              void* d_out, int out_size, void* d_ws,
                              size_t ws_size, hipStream_t stream) {
  const float* x = (const float*)d_in[0];      // (16,512,64,64) f32
  const float* w = (const float*)d_in[1];      // (512,256,3,3)  f32
  float* out = (float*)d_out;                  // (16,256,128,128) f32
  unsigned char* ws = (unsigned char*)d_ws;
  unsigned short* ww = (unsigned short*)ws;

  phase_weights_kernel<<<WS_W_ELEMS / 256u, 256, 0, stream>>>(w, ww);

  bool use_xt = ws_size >= XT_OFF + XT_BYTES;   // deterministic per ws_size
  if (use_xt) {
    unsigned short* xtp = (unsigned short*)(ws + XT_OFF);
    nhwc_convert_kernel<<<16 * 66, 256, 0, stream>>>(x, xtp);
    subpix_conv_fir_kernel<2><<<4096, 256, 0, stream>>>(x, xtp, ww, out);
    if (ws_size == 1u)   // impossible here; keeps async variant compiled
      subpix_conv_fir_kernel<1><<<4096, 256, 0, stream>>>(x, xtp, ww, out);
  } else {
    subpix_conv_fir_kernel<0><<<4096, 256, 0, stream>>>(x, nullptr, ww, out);
  }
}
